// GraphAttention_3959959847393
// MI455X (gfx1250) — compile-verified, matrix operand order checked by reading, not run on hardware
//
#include <hip/hip_runtime.h>

#define GN 8192
#define GD 256
#define NCH (GN / 32)

typedef __attribute__((ext_vector_type(16))) __bf16 v16bf;
typedef __attribute__((ext_vector_type(8)))  float  v8f;
typedef __attribute__((ext_vector_type(4)))  float  v4f;

union BF16x16 { uint4 q[2]; v16bf v; };

__device__ __forceinline__ unsigned short f2bf(float f) {
  unsigned u = __float_as_uint(f);
  u += 0x7fffu + ((u >> 16) & 1u);          // round-to-nearest-even
  return (unsigned short)(u >> 16);
}
__device__ __forceinline__ unsigned pack2bf(float a, float b) {
  return (unsigned)f2bf(a) | ((unsigned)f2bf(b) << 16);
}

// CDNA5 async copy: 16B global -> LDS, tracked by ASYNCcnt (no VGPR landing zone)
__device__ __forceinline__ void async_g2l_b128(unsigned lds_off, const void* gaddr) {
  asm volatile("global_load_async_to_lds_b128 %0, %1, off"
               :: "v"(lds_off), "v"((unsigned long long)gaddr) : "memory");
}

// ---------------------------------------------------------------- conversions
__global__ __launch_bounds__(256) void k_cvt_bf16(const float* __restrict__ s,
                                                  unsigned short* __restrict__ d,
                                                  int n) {
  int i = blockIdx.x * 256 + threadIdx.x;
  if (i < n) d[i] = f2bf(s[i]);
}

// ------------------------------------------------- h = x @ W^T  (WMMA bf16)
// One 16x16 output tile per wave; also emits transposed bf16 copy hbT[GD][GN]
// so that pass-2 B tiles are K-contiguous (two b128 loads per lane per tile).
__global__ __launch_bounds__(128) void k_h_gemm(const unsigned short* __restrict__ xb,
                                                const unsigned short* __restrict__ Wb,
                                                float* __restrict__ h,
                                                unsigned short* __restrict__ hbT) {
  const int lane = threadIdx.x & 31;
  const int wave = threadIdx.x >> 5;
  const int tile = blockIdx.x * 4 + wave;      // 8192 tiles = 512 x 16
  const int i0   = (tile >> 4) * 16;
  const int n0   = (tile & 15) * 16;
  const int mn   = lane & 15;                  // A row / B col / D col
  const int half = lane >> 4;                  // K-group select

  v8f acc = {};
  for (int k0 = 0; k0 < GD; k0 += 32) {
    BF16x16 A, B;
    const unsigned short* ap = xb + (size_t)(i0 + mn) * GD + k0 + half * 8;
    A.q[0] = *(const uint4*)(ap);
    A.q[1] = *(const uint4*)(ap + 16);
    const unsigned short* bp = Wb + (size_t)(n0 + mn) * GD + k0 + half * 8;
    B.q[0] = *(const uint4*)(bp);
    B.q[1] = *(const uint4*)(bp + 16);
    acc = __builtin_amdgcn_wmma_f32_16x16x32_bf16(false, A.v, false, B.v,
                                                  (short)0, acc, false, false);
  }
#pragma unroll
  for (int r = 0; r < 8; ++r) {
    int row = i0 + r + half * 8;
    int col = n0 + mn;
    float v = acc[r];
    h[(size_t)row * GD + col]  = v;
    hbT[(size_t)col * GN + row] = f2bf(v);
  }
}

// ------------------------------------------------------- s1 = h@a1, s2 = h@a2
__global__ __launch_bounds__(256) void k_scores(const float* __restrict__ h,
                                                const float* __restrict__ a,
                                                float* __restrict__ s1,
                                                float* __restrict__ s2) {
  int i = blockIdx.x * 256 + threadIdx.x;
  const float* hr = h + (size_t)i * GD;
  float t1 = 0.f, t2 = 0.f;
#pragma unroll 8
  for (int c = 0; c < GD; ++c) {
    float hv = hr[c];
    t1 = fmaf(hv, a[c], t1);
    t2 = fmaf(hv, a[GD + c], t2);
  }
  s1[i] = t1; s2[i] = t2;
}

// ---------------------------------------- pass 1: per-row softmax max & sum
__global__ __launch_bounds__(256) void k_pass1(const float* __restrict__ adj,
                                               const float* __restrict__ s1,
                                               const float* __restrict__ s2,
                                               float* __restrict__ rowm,
                                               float* __restrict__ rowl) {
  __shared__ float sm[256], sl[256];
  const int i   = blockIdx.x;
  const int tid = threadIdx.x;
  const float s1i = s1[i];
  float mx = -3.0e38f, sum = 0.f;
  const float* arow = adj + (size_t)i * GN;
  for (int j = tid; j < GN; j += 256) {
    // streaming read: non-temporal, keep L2 for hbT/P traffic
    float conn = __builtin_nontemporal_load(arow + j) + ((j == i) ? 1.f : 0.f);
    if (conn != 0.f) {
      float t = s1i + s2[j];
      float e = (t > 0.f) ? t : 0.2f * t;
      if (e > mx) { sum *= __expf(mx - e); mx = e; }
      sum += __expf(e - mx);
    }
  }
  sm[tid] = mx; sl[tid] = sum;
  __syncthreads();
  for (int s = 128; s > 0; s >>= 1) {
    if (tid < s) {
      float m1 = sm[tid], l1 = sl[tid];
      float m2 = sm[tid + s], l2 = sl[tid + s];
      float M = fmaxf(m1, m2);
      sm[tid] = M;
      sl[tid] = l1 * __expf(m1 - M) + l2 * __expf(m2 - M);
    }
    __syncthreads();
  }
  if (tid == 0) { rowm[i] = sm[0]; rowl[i] = sl[0]; }
}

// ------- pass 2: fused mask -> exp/l -> G store -> WMMA accumulate out=ELU(G@h)
// 128 threads (4 waves) per 16-row block; wave w owns output cols [w*64, w*64+64).
// adj tiles are double-buffered in LDS via global_load_async_to_lds_b128
// (ASYNCcnt) so HBM latency overlaps the exp/WMMA work on the previous tile.
__global__ __launch_bounds__(128) void k_pass2(const float* __restrict__ adj,
                                               const float* __restrict__ s1,
                                               const float* __restrict__ s2,
                                               const float* __restrict__ rowm,
                                               const float* __restrict__ rowl,
                                               const unsigned short* __restrict__ hbT,
                                               float* __restrict__ out,
                                               float* __restrict__ G) {
  __shared__ __align__(16) float    adjS[2][16 * 32];  // double-buffered adj tile
  __shared__ __align__(16) unsigned pTu[16 * 16];      // 16x32 bf16 P tile
  __shared__ float sS1[16], sM[16], sIL[16];

  const int tid  = threadIdx.x;
  const int lane = tid & 31;
  const int wave = tid >> 5;
  const int i0   = blockIdx.x * 16;
  const int mn   = lane & 15;
  const int half = lane >> 4;
  const int nBase = wave * 64;
  const int prow = tid >> 3;          // 0..15
  const int pcol = (tid & 7) * 4;     // 0..28 step 4

  if (tid < 16) {
    sS1[tid] = s1[i0 + tid];
    sM[tid]  = rowm[i0 + tid];
    sIL[tid] = 1.0f / rowl[i0 + tid];
  }

  const float* arow = adj + (size_t)(i0 + prow) * GN + pcol;  // this thread's slot
  float* myAdj[2];
  unsigned myOff[2];
  myAdj[0] = &adjS[0][prow * 32 + pcol];
  myAdj[1] = &adjS[1][prow * 32 + pcol];
  myOff[0] = (unsigned)(size_t)myAdj[0];
  myOff[1] = (unsigned)(size_t)myAdj[1];

  // prologue: start chunk 0 transfer
  async_g2l_b128(myOff[0], arow);

  v8f acc[4] = {{}, {}, {}, {}};
  __syncthreads();  // sS1/sM/sIL visible

  for (int ch = 0; ch < NCH; ++ch) {
    const int j0  = ch * 32;
    const int cur = ch & 1;

    // kick off next tile's async copy, then retire the current one
    if (ch + 1 < NCH) {
      async_g2l_b128(myOff[cur ^ 1], arow + (ch + 1) * 32);
      asm volatile("s_wait_asynccnt 0x1" ::: "memory");
    } else {
      asm volatile("s_wait_asynccnt 0x0" ::: "memory");
    }

    // each thread consumes exactly the 16B slot it transferred
    v4f av = *(const v4f*)myAdj[cur];
    const float s1i = sS1[prow], mi = sM[prow], il = sIL[prow];
    const v4f s2v = *(const v4f*)(s2 + j0 + pcol);
    v4f p;
#pragma unroll
    for (int q = 0; q < 4; ++q) {
      int j = j0 + pcol + q;
      float conn = av[q] + ((j == i0 + prow) ? 1.f : 0.f);
      float t = s1i + s2v[q];
      float e = (t > 0.f) ? t : 0.2f * t;
      p[q] = (conn != 0.f) ? __expf(e - mi) * il : 0.f;
    }
    // normalized attention row -> G: write-once stream, bypass L2 residency
    __builtin_nontemporal_store(p, (v4f*)(G + (size_t)(i0 + prow) * GN + j0 + pcol));
    // bf16 pack into LDS for WMMA A operand
    pTu[prow * 16 + (pcol >> 1)]     = pack2bf(p[0], p[1]);
    pTu[prow * 16 + (pcol >> 1) + 1] = pack2bf(p[2], p[3]);
    __syncthreads();

    BF16x16 A;
    A.q[0] = *(const uint4*)&pTu[mn * 16 + half * 4];
    A.q[1] = *(const uint4*)&pTu[mn * 16 + 8 + half * 4];
#pragma unroll
    for (int t = 0; t < 4; ++t) {
      BF16x16 B;
      const unsigned short* bp =
          hbT + (size_t)(nBase + t * 16 + mn) * GN + j0 + half * 8;
      B.q[0] = *(const uint4*)(bp);
      B.q[1] = *(const uint4*)(bp + 16);
      acc[t] = __builtin_amdgcn_wmma_f32_16x16x32_bf16(false, A.v, false, B.v,
                                                       (short)0, acc[t], false, false);
    }
    __syncthreads();   // protect pTu before next chunk overwrites it
  }

  // epilogue: ELU and store out[16 x 256] block
#pragma unroll
  for (int t = 0; t < 4; ++t) {
#pragma unroll
    for (int r = 0; r < 8; ++r) {
      int row = i0 + r + half * 8;
      int col = nBase + t * 16 + mn;
      float v = acc[t][r];
      out[(size_t)row * GD + col] = (v > 0.f) ? v : (__expf(v) - 1.f);
    }
  }
}

// ---------------------------------------------------------------- launcher
extern "C" void kernel_launch(void* const* d_in, const int* in_sizes, int n_in,
                              void* d_out, int out_size, void* d_ws, size_t ws_size,
                              hipStream_t stream) {
  const float* x   = (const float*)d_in[0];
  const float* adj = (const float*)d_in[1];
  const float* W   = (const float*)d_in[2];
  const float* a   = (const float*)d_in[3];

  float* out = (float*)d_out;                 // [GN, GD]
  float* G   = out + (size_t)GN * GD;         // [GN, GN]

  char* ws = (char*)d_ws;
  unsigned short* xb  = (unsigned short*)(ws);                  // 4 MB
  unsigned short* Wb  = (unsigned short*)(ws + 4194304);        // 128 KB
  float*          h   = (float*)(ws + 4325376);                 // 8 MB
  unsigned short* hbT = (unsigned short*)(ws + 12713984);       // 4 MB
  float*          s1  = (float*)(ws + 16908288);                // 32 KB
  float*          s2  = (float*)(ws + 16941056);                // 32 KB
  float*          rm  = (float*)(ws + 16973824);                // 32 KB
  float*          rl  = (float*)(ws + 17006592);                // 32 KB

  k_cvt_bf16<<<(GN * GD + 255) / 256, 256, 0, stream>>>(x, xb, GN * GD);
  k_cvt_bf16<<<(GD * GD + 255) / 256, 256, 0, stream>>>(W, Wb, GD * GD);
  k_h_gemm  <<<(GN / 16) * (GD / 16) / 4, 128, 0, stream>>>(xb, Wb, h, hbT);
  k_scores  <<<GN / 256, 256, 0, stream>>>(h, a, s1, s2);
  k_pass1   <<<GN, 256, 0, stream>>>(adj, s1, s2, rm, rl);
  k_pass2   <<<GN / 16, 128, 0, stream>>>(adj, s1, s2, rm, rl, hbT, out, G);
}
